// MixtureOfExperts_72387378807203
// MI455X (gfx1250) — compile-verified, matrix operand order checked by reading, not run on hardware
//
#include <hip/hip_runtime.h>
#include <hip/hip_bf16.h>

// ---------------- Problem constants ----------------
#define E_  8
#define K_  2
#define H_  1024
#define F_  4096
#define B_  2
#define S_  2048
#define T_  (B_ * S_)              // 4096 tokens
#define NROW_EXP (T_ * K_)         // 8192 gathered expert rows (exact)
#define NROW_ALL (NROW_EXP + T_)   // + shared-expert rows
#define TILES_PER_SEG (T_ / 128)   // worst case: all tokens on one expert -> 32 tiles
#define NSEG (E_ + 1)              // 8 experts + shared expert

typedef __attribute__((ext_vector_type(16))) __bf16 v16bf;
typedef __attribute__((ext_vector_type(8)))  float  v8f;

union Frag16 { uint4 q[2]; v16bf v; };   // 32 bytes: per-lane 16 bf16 = two 16B chunks

__device__ __forceinline__ unsigned short f2bf(float f) {
  unsigned int u = __float_as_uint(f);
  unsigned int r = u + 0x7FFFu + ((u >> 16) & 1u);   // round-to-nearest-even
  return (unsigned short)(r >> 16);
}
__device__ __forceinline__ float gelu_exact(float x) {
  return 0.5f * x * (1.0f + erff(x * 0.70710678118654752440f));
}

// ---- CDNA5 async global->LDS (ASYNCcnt) helpers ----
// Generic pointers into LDS carry the LDS byte offset in their low 32 bits
// that value is what the async
// instruction's VDST operand wants.
__device__ __forceinline__ unsigned lds_addr_of(const void* p) {
  return (unsigned)(unsigned long long)p;
}
__device__ __forceinline__ void async_load_b128(unsigned lds_addr, const void* gaddr) {
  asm volatile("global_load_async_to_lds_b128 %0, %1, off"
               :: "v"(lds_addr), "v"(gaddr) : "memory");
}
template <int N>
__device__ __forceinline__ void wait_asynccnt() {
  asm volatile("s_wait_asynccnt %0" :: "i"(N) : "memory");
}

// ---------------- 0) fp32 -> bf16 elementwise ----------------
__global__ void cvt_bf16_kernel(const float* __restrict__ in,
                                unsigned short* __restrict__ out, int n) {
  int i = blockIdx.x * blockDim.x + threadIdx.x;
  if (i < n) out[i] = f2bf(in[i]);
}

// ---------------- 1) fp32 [R,C] -> bf16 [C,R] (batched over z) ----------------
__global__ void transpose_cvt_kernel(const float* __restrict__ in,
                                     unsigned short* __restrict__ out,
                                     int R, int C) {
  __shared__ float tile[32][33];
  size_t bo = (size_t)blockIdx.z * (size_t)R * (size_t)C;
  const float* inb = in + bo;
  unsigned short* outb = out + bo;
  int c0 = blockIdx.x * 32, r0 = blockIdx.y * 32;
  int tx = threadIdx.x, ty = threadIdx.y;     // block (32, 8)
#pragma unroll
  for (int j = 0; j < 4; ++j)
    tile[ty + j * 8][tx] = inb[(size_t)(r0 + ty + j * 8) * C + (c0 + tx)];
  __syncthreads();
#pragma unroll
  for (int j = 0; j < 4; ++j)
    outb[(size_t)(c0 + ty + j * 8) * R + (r0 + tx)] = f2bf(tile[tx][ty + j * 8]);
}

// ---------------- 2) zero small counters ----------------
__global__ void zero_ints_kernel(int* p, int n) {
  int i = blockIdx.x * blockDim.x + threadIdx.x;
  if (i < n) p[i] = 0;
}

// ---------------- 3) router: logits -> softmax -> top2 ----------------
__global__ void router_topk_kernel(const float* __restrict__ x,
                                   const float* __restrict__ rw,
                                   const float* __restrict__ rb,
                                   int* __restrict__ top2i,
                                   float* __restrict__ top2w,
                                   int* __restrict__ counts) {
  int t = blockIdx.x * blockDim.x + threadIdx.x;
  if (t >= T_) return;
  float acc[E_];
#pragma unroll
  for (int e = 0; e < E_; ++e) acc[e] = rb[e];
  const float* xr = x + (size_t)t * H_;
  for (int h = 0; h < H_; ++h) {
    float xv = xr[h];
    const float* w = rw + (size_t)h * E_;
#pragma unroll
    for (int e = 0; e < E_; ++e) acc[e] += xv * w[e];
  }
  float mx = acc[0];
#pragma unroll
  for (int e = 1; e < E_; ++e) mx = fmaxf(mx, acc[e]);
  float p[E_]; float s = 0.f;
#pragma unroll
  for (int e = 0; e < E_; ++e) { p[e] = expf(acc[e] - mx); s += p[e]; }
  float inv = 1.f / s;
#pragma unroll
  for (int e = 0; e < E_; ++e) p[e] *= inv;
  // top-1 (lowest index on ties, like jax top_k)
  int i0 = 0; float w0 = p[0];
#pragma unroll
  for (int e = 1; e < E_; ++e) if (p[e] > w0) { w0 = p[e]; i0 = e; }
  // top-2
  int i1 = -1; float w1 = -1.f;
#pragma unroll
  for (int e = 0; e < E_; ++e) if (e != i0 && p[e] > w1) { w1 = p[e]; i1 = e; }
  top2i[t * 2 + 0] = i0;  top2w[t * 2 + 0] = w0;
  top2i[t * 2 + 1] = i1;  top2w[t * 2 + 1] = w1;
  atomicAdd(&counts[i0], 1);
  atomicAdd(&counts[i1], 1);
}

// ---------------- 4) exclusive scan over 8 experts ----------------
__global__ void scan_offsets_kernel(const int* __restrict__ counts,
                                    int* __restrict__ offsets,
                                    int* __restrict__ counts2) {
  if (threadIdx.x == 0) {
    int s = 0;
    for (int e = 0; e < E_; ++e) { offsets[e] = s; s += counts[e]; counts2[e] = 0; }
  }
}

// ---------------- 5) build gathered row lists ----------------
__global__ void fill_rows_kernel(const int* __restrict__ top2i,
                                 const int* __restrict__ offsets,
                                 int* __restrict__ counts2,
                                 int* __restrict__ row_token,
                                 int* __restrict__ rowloc) {
  int t = blockIdx.x * blockDim.x + threadIdx.x;
  if (t >= T_) return;
#pragma unroll
  for (int k = 0; k < K_; ++k) {
    int e = top2i[t * 2 + k];
    int pos = atomicAdd(&counts2[e], 1);
    int r = offsets[e] + pos;
    row_token[r] = t;
    rowloc[t * 2 + k] = r;
  }
}

// ---------------- 6) grouped GEMM (bf16 WMMA, async double-buffered) ----------
// Block tile 128x128x32, 256 threads (8 wave32), wave tile 64x32 (4x2 WMMA).
// A is row-gathered; B is pre-transposed to [N][K] so both fragments are
// contiguous-K per lane, matching the CDNA5 16-bit A/B VGPR layout:
//   lanes 0-15 : K 0..7 and 16..23 ; lanes 16-31 : K 8..15 and 24..31.
// Tiles are staged with GLOBAL_LOAD_ASYNC_TO_LDS_B128 (ASYNCcnt), double
// buffered: issue tile k+1, s_wait_asynccnt 4 -> tile k landed (async loads
// complete in order), barrier, compute.
template <bool UP>
__global__ __launch_bounds__(256)
void moe_gemm_kernel(const unsigned short* __restrict__ Abase,   // UP: xb[T,H] ; DOWN: act[NROW_ALL,F]
                     const unsigned short* __restrict__ BT_exp,  // [E][N][K]
                     const unsigned short* __restrict__ BT_sh,   // [N][K]
                     const float* __restrict__ bias_exp,         // UP: up_b[E,F]
                     const float* __restrict__ bias_sh,          // UP: sh_up_b[F]
                     const int* __restrict__ counts,
                     const int* __restrict__ offsets,
                     const int* __restrict__ row_token,
                     unsigned short* __restrict__ actOut,        // UP out: [NROW_ALL,F] bf16
                     float* __restrict__ downOut) {              // DOWN out: [NROW_ALL,H] f32
  constexpr int Kdim = UP ? H_ : F_;
  constexpr int Nout = UP ? F_ : H_;
  constexpr int LDS_STRIDE = 48;           // halves; 96B rows keep 16B alignment
  constexpr int HALF_PER_MAT = 128 * LDS_STRIDE;   // 6144 halves = 12 KB
  constexpr int HALF_PER_BUF = 2 * HALF_PER_MAT;   // A + B
  constexpr int NT = Kdim / 32;            // K tiles

  const int vt = blockIdx.y;                 // virtual row tile 0..NSEG*32-1
  const int seg = vt / TILES_PER_SEG;        // expert 0..7, or 8 == shared
  const int tileOff = (vt % TILES_PER_SEG) * 128;
  int rowsInSeg, rowBase;
  if (seg < E_) { rowsInSeg = counts[seg]; rowBase = offsets[seg]; }
  else          { rowsInSeg = T_;          rowBase = NROW_EXP;     }
  if (tileOff >= rowsInSeg) return;

  const unsigned short* BT =
      (seg < E_) ? (BT_exp + (size_t)seg * (size_t)Nout * Kdim) : BT_sh;
  const int n0 = blockIdx.x * 128;

  __shared__ __align__(16) unsigned short sm[2 * HALF_PER_BUF];  // 48 KB

  const int tid  = threadIdx.x;
  const int lane = tid & 31, wv = tid >> 5;
  const int wm = wv >> 2, wn = wv & 3;       // 2x4 wave grid
  const int g  = lane >> 4, lm = lane & 15;  // half-wave group / lane-in-group

  // --- per-thread staging sources (32B of A + 32B of B per tile) ---
  const int lrA  = tid >> 1;                 // 0..127 row within tile
  const int segA = (tid & 1) * 16;           // which 16-half chunk of K-slice
  const int mIdx = tileOff + lrA;
  const int mCl  = (mIdx < rowsInSeg) ? mIdx : (rowsInSeg - 1);
  const unsigned short* Arow;
  if constexpr (UP) {
    int tok = (seg < E_) ? row_token[rowBase + mCl] : mCl;
    Arow = Abase + (size_t)tok * Kdim;
  } else {
    Arow = Abase + (size_t)(rowBase + mCl) * Kdim;
  }
  const unsigned short* Brow = BT + (size_t)(n0 + lrA) * Kdim;

  // per-thread LDS destinations (byte addresses) for each buffer
  unsigned dstA[2], dstB[2];
#pragma unroll
  for (int d = 0; d < 2; ++d) {
    dstA[d] = lds_addr_of(sm + d * HALF_PER_BUF + lrA * LDS_STRIDE + segA);
    dstB[d] = lds_addr_of(sm + d * HALF_PER_BUF + HALF_PER_MAT + lrA * LDS_STRIDE + segA);
  }
  auto issue_tile = [&](int kt, int d) {
    const unsigned short* ga = Arow + kt * 32 + segA;
    const unsigned short* gb = Brow + kt * 32 + segA;
    async_load_b128(dstA[d],      ga);
    async_load_b128(dstA[d] + 16, ga + 8);
    async_load_b128(dstB[d],      gb);
    async_load_b128(dstB[d] + 16, gb + 8);
  };

  v8f acc[4][2];
#pragma unroll
  for (int mt = 0; mt < 4; ++mt)
#pragma unroll
    for (int nt = 0; nt < 2; ++nt) acc[mt][nt] = 0;

  issue_tile(0, 0);
  for (int kt = 0; kt < NT; ++kt) {
    const int d = kt & 1;
    if (kt + 1 < NT) {
      issue_tile(kt + 1, d ^ 1);
      wait_asynccnt<4>();       // tile kt landed; tile kt+1 still in flight
    } else {
      wait_asynccnt<0>();
    }
    __syncthreads();

    const unsigned short* Abuf = sm + d * HALF_PER_BUF;
    const unsigned short* Bbuf = Abuf + HALF_PER_MAT;

    Frag16 b[2];
#pragma unroll
    for (int nt = 0; nt < 2; ++nt) {
      int nrow = wn * 32 + nt * 16 + lm;
      b[nt].q[0] = *(const uint4*)(Bbuf + nrow * LDS_STRIDE + 8 * g);
      b[nt].q[1] = *(const uint4*)(Bbuf + nrow * LDS_STRIDE + 16 + 8 * g);
    }
#pragma unroll
    for (int mt = 0; mt < 4; ++mt) {
      Frag16 a;
      int mrow = wm * 64 + mt * 16 + lm;
      a.q[0] = *(const uint4*)(Abuf + mrow * LDS_STRIDE + 8 * g);
      a.q[1] = *(const uint4*)(Abuf + mrow * LDS_STRIDE + 16 + 8 * g);
#pragma unroll
      for (int nt = 0; nt < 2; ++nt) {
        acc[mt][nt] = __builtin_amdgcn_wmma_f32_16x16x32_bf16(
            false, a.v, false, b[nt].v, (short)0, acc[mt][nt], false, false);
      }
    }
    __syncthreads();   // forces s_wait_dscnt 0 before next buffer reuse
  }

  // C/D layout: VGPR i -> M = i + 8*g, N = lm (within 16x16 subtile)
#pragma unroll
  for (int mt = 0; mt < 4; ++mt)
#pragma unroll
    for (int nt = 0; nt < 2; ++nt)
#pragma unroll
      for (int i = 0; i < 8; ++i) {
        int mloc = wm * 64 + mt * 16 + i + 8 * g;
        if (tileOff + mloc < rowsInSeg) {
          int n = n0 + wn * 32 + nt * 16 + lm;
          size_t r = (size_t)(rowBase + tileOff + mloc);
          float v = acc[mt][nt][i];
          if constexpr (UP) {
            float bias = (seg < E_) ? bias_exp[(size_t)seg * F_ + n] : bias_sh[n];
            actOut[r * F_ + n] = f2bf(gelu_exact(v + bias));
          } else {
            downOut[r * H_ + n] = v;
          }
        }
      }
}

// ---------------- 7) final combine ----------------
__global__ void combine_kernel(const float* __restrict__ x,
                               const float* __restrict__ down_b,
                               const float* __restrict__ sh_down_b,
                               const int* __restrict__ top2i,
                               const float* __restrict__ top2w,
                               const int* __restrict__ rowloc,
                               const float* __restrict__ downOut,
                               float* __restrict__ out) {
  int idx = blockIdx.x * blockDim.x + threadIdx.x;
  if (idx >= T_ * H_) return;
  int t = idx / H_, h = idx - t * H_;
  float v = x[idx] + downOut[(size_t)(NROW_EXP + t) * H_ + h] + sh_down_b[h];
#pragma unroll
  for (int k = 0; k < K_; ++k) {
    int e = top2i[t * 2 + k];
    float w = top2w[t * 2 + k];
    int r = rowloc[t * 2 + k];
    v += w * (downOut[(size_t)r * H_ + h] + down_b[(size_t)e * H_ + h]);
  }
  out[idx] = v;
}

// ---------------- host side ----------------
extern "C" void kernel_launch(void* const* d_in, const int* in_sizes, int n_in,
                              void* d_out, int out_size, void* d_ws, size_t ws_size,
                              hipStream_t stream) {
  const float* x         = (const float*)d_in[0];
  const float* router_w  = (const float*)d_in[1];
  const float* router_b  = (const float*)d_in[2];
  const float* up_w      = (const float*)d_in[3];
  const float* up_b      = (const float*)d_in[4];
  const float* down_w    = (const float*)d_in[5];
  const float* down_b    = (const float*)d_in[6];
  const float* sh_up_w   = (const float*)d_in[7];
  const float* sh_up_b   = (const float*)d_in[8];
  const float* sh_down_w = (const float*)d_in[9];
  const float* sh_down_b = (const float*)d_in[10];
  float* out = (float*)d_out;

  // workspace carve-out (~311 MB total)
  char* base = (char*)d_ws;
  size_t o = 0;
  auto carve = [&](size_t bytes) -> char* {
    char* r = base + o;
    o = (o + bytes + 255) & ~(size_t)255;
    return r;
  };
  unsigned short* xb      = (unsigned short*)carve((size_t)T_ * H_ * 2);
  unsigned short* upT     = (unsigned short*)carve((size_t)E_ * F_ * H_ * 2);
  unsigned short* downT   = (unsigned short*)carve((size_t)E_ * H_ * F_ * 2);
  unsigned short* shUpT   = (unsigned short*)carve((size_t)F_ * H_ * 2);
  unsigned short* shDownT = (unsigned short*)carve((size_t)H_ * F_ * 2);
  unsigned short* act     = (unsigned short*)carve((size_t)NROW_ALL * F_ * 2);
  float*          dOut    = (float*)carve((size_t)NROW_ALL * H_ * 4);
  int*   top2i     = (int*)carve((size_t)T_ * K_ * 4);
  float* top2w     = (float*)carve((size_t)T_ * K_ * 4);
  int*   rowloc    = (int*)carve((size_t)T_ * K_ * 4);
  int*   row_token = (int*)carve((size_t)NROW_EXP * 4);
  int*   counts    = (int*)carve(E_ * 4);
  int*   offsets   = (int*)carve(E_ * 4);
  int*   counts2   = (int*)carve(E_ * 4);
  (void)ws_size; (void)n_in; (void)in_sizes; (void)out_size;

  // 0) precision staging: x -> bf16 ; weights -> bf16 transposed [N][K]
  {
    int n = T_ * H_;
    cvt_bf16_kernel<<<(n + 255) / 256, 256, 0, stream>>>(x, xb, n);
  }
  transpose_cvt_kernel<<<dim3(F_ / 32, H_ / 32, E_), dim3(32, 8), 0, stream>>>(up_w, upT, H_, F_);
  transpose_cvt_kernel<<<dim3(H_ / 32, F_ / 32, E_), dim3(32, 8), 0, stream>>>(down_w, downT, F_, H_);
  transpose_cvt_kernel<<<dim3(F_ / 32, H_ / 32, 1), dim3(32, 8), 0, stream>>>(sh_up_w, shUpT, H_, F_);
  transpose_cvt_kernel<<<dim3(H_ / 32, F_ / 32, 1), dim3(32, 8), 0, stream>>>(sh_down_w, shDownT, F_, H_);

  // 1) routing
  zero_ints_kernel<<<1, 32, 0, stream>>>(counts, E_);
  router_topk_kernel<<<T_ / 256, 256, 0, stream>>>(x, router_w, router_b, top2i, top2w, counts);
  scan_offsets_kernel<<<1, 32, 0, stream>>>(counts, offsets, counts2);
  fill_rows_kernel<<<T_ / 256, 256, 0, stream>>>(top2i, offsets, counts2, row_token, rowloc);

  // 2) grouped up-GEMM (+gelu) and down-GEMM over 8 experts + shared segment
  moe_gemm_kernel<true><<<dim3(F_ / 128, NSEG * TILES_PER_SEG), 256, 0, stream>>>(
      xb, upT, shUpT, up_b, sh_up_b, counts, offsets, row_token, act, dOut);
  moe_gemm_kernel<false><<<dim3(H_ / 128, NSEG * TILES_PER_SEG), 256, 0, stream>>>(
      act, downT, shDownT, nullptr, nullptr, counts, offsets, row_token, act, dOut);

  // 3) combine: x + shared + weighted expert rows (+biases)
  combine_kernel<<<(T_ * H_ + 255) / 256, 256, 0, stream>>>(
      x, down_b, sh_down_b, top2i, top2w, rowloc, dOut, out);
}
